// NKATDiracKAN_60945585930798
// MI455X (gfx1250) — matrix operand order repaired and checked by reading, not compiled, not run on hardware
//
#include <hip/hip_runtime.h>

typedef __attribute__((ext_vector_type(16))) _Float16 v16h;
typedef __attribute__((ext_vector_type(8)))  float    v8f;

// integral-constant tag + constexpr-for: loop indices become template constants.
template <int V> struct ic { static constexpr int value = V; };

template <int N, typename F>
__device__ __forceinline__ void static_for(F&& f) {
    if constexpr (N > 0) {
        static_for<N - 1>(f);
        f(ic<N - 1>{});
    }
}

#define EXP2_NEG5_LOG2E  (-7.2134752044448170f)   // -5 * log2(e)
#define EXP2_NEG2_LOG2E  (-2.8853900817779268f)   // -2 * log2(e)

// Branch-free tanh via raw v_exp_f32 (base-2): copysign((1-t)/(1+t), x),
// t = 2^(-2*log2e*|x|). One transcendental + one rcp, no EXEC divergence.
__device__ __forceinline__ float fast_tanh(float x) {
    float ax = __builtin_fabsf(x);
    float t  = __builtin_amdgcn_exp2f(EXP2_NEG2_LOG2E * ax);
    float r  = (1.0f - t) * __builtin_amdgcn_rcpf(1.0f + t);
    return __builtin_copysignf(r, x);
}

// ---------------------------------------------------------------------------
// One A-fragment slot. kk = KK0 (lane half 0) or KK0+8 (half 1); both are
// compile-time, so i = kk/5 and the grid center fold to constants with at
// most one v_cndmask each (no runtime division by 5).
// ---------------------------------------------------------------------------
template <int DIN, int INSTRIDE, int KK0>
__device__ __forceinline__ float basis_slot(const float* hin, int m, int half) {
    constexpr int  KK1 = KK0 + 8;
    constexpr bool V0  = KK0 < DIN * 5;
    constexpr bool V1  = KK1 < DIN * 5;
    if constexpr (!V0 && !V1) {
        return 0.0f;
    } else {
        constexpr int I0 = V0 ? KK0 / 5 : KK1 / 5;
        constexpr int I1 = V1 ? KK1 / 5 : I0;
        constexpr int G0 = V0 ? KK0 % 5 : KK1 % 5;
        constexpr int G1 = V1 ? KK1 % 5 : G0;
        int   i = (I0 == I1) ? I0 : (half ? I1 : I0);
        float c = (G0 == G1) ? (-1.0f + 0.5f * (float)G0)
                             : (half ? (-1.0f + 0.5f * (float)G1)
                                     : (-1.0f + 0.5f * (float)G0));
        float xn = hin[m * INSTRIDE + i];
        float d  = xn - c;
        float e  = __builtin_amdgcn_exp2f(EXP2_NEG5_LOG2E * (d * d));
        if constexpr (V0 && V1) return e;
        else                    return (half ? V1 : V0) ? e : 0.0f;
    }
}

// A-fragment (16x32 f16, ISA 7.12.2): half = lane>>4, row m = lane&15,
// slot t -> K = KS*32 + half*8 + (t<8 ? t : t+8).
template <int DIN, int INSTRIDE, int KS>
__device__ __forceinline__ v16h build_a(const float* hin, int m, int half) {
    v16h a;
    static_for<16>([&](auto tc) {
        constexpr int T   = decltype(tc)::value;
        constexpr int KK0 = KS * 32 + (T < 8 ? T : T + 8);
        a[T] = (_Float16)basis_slot<DIN, INSTRIDE, KK0>(hin, m, half);
    });
    return a;
}

// ---------------------------------------------------------------------------
// Weight prep (runs once, 1 block): fold scale into coeffs, cast to f16, and
// store in WMMA B-fragment order for V_WMMA_F32_16X16X32_F16:
//   lane l holds column n = l&15, K = kstep*32 + (l>>4)*8 + (t<8 ? t : t+8).
// ---------------------------------------------------------------------------
template <int DIN, int DOUT, int KSTEPS, int NTILES>
__device__ __forceinline__ void prep_weights(const float* __restrict__ c,
                                             const float* __restrict__ s,
                                             _Float16* wdst, int tid, int nthreads) {
    const int total = NTILES * KSTEPS * 512;
    for (int idx = tid; idx < total; idx += nthreads) {
        int ntile = idx / (KSTEPS * 512);
        int r     = idx - ntile * (KSTEPS * 512);
        int kstep = r >> 9;
        int q     = r & 511;
        int ln    = q >> 4;
        int t     = q & 15;
        int K = kstep * 32 + ((ln >> 4) << 3) + (t < 8 ? t : t + 8);
        int j = ntile * 16 + (ln & 15);
        float v = 0.0f;
        if (K < DIN * 5 && j < DOUT) {
            int i = K / 5;
            int g = K - i * 5;
            v = c[(i * DOUT + j) * 5 + g] * s[i * DOUT + j];
        }
        wdst[idx] = (_Float16)v;
    }
}

template <int DIN, int DOUT>
__device__ __forceinline__ void prep_bias(const float* __restrict__ shift,
                                          float* bdst, int tid, int nthreads) {
    for (int j = tid; j < DOUT; j += nthreads) {
        float acc = 0.0f;
        for (int i = 0; i < DIN; ++i) acc += shift[i * DOUT + j];
        bdst[j] = acc;
    }
}

// ws layout: [0, 32768) f16 weights in fragment order; [32768, 33280) f32 bias
__global__ void __launch_bounds__(256)
nkat_prep_kernel(const float* __restrict__ cf0, const float* __restrict__ sc0, const float* __restrict__ sf0,
                 const float* __restrict__ cf1, const float* __restrict__ sc1, const float* __restrict__ sf1,
                 const float* __restrict__ cf2, const float* __restrict__ sc2, const float* __restrict__ sf2,
                 const float* __restrict__ cf3, const float* __restrict__ sc3, const float* __restrict__ sf3,
                 _Float16* __restrict__ wOut, float* __restrict__ bOut) {
    const int tid = threadIdx.x;
    prep_weights<4, 64, 1, 4>(cf0, sc0, wOut + 0,     tid, 256);
    prep_weights<64, 32, 10, 2>(cf1, sc1, wOut + 2048,  tid, 256);
    prep_weights<32, 16, 5, 1>(cf2, sc2, wOut + 12288, tid, 256);
    prep_weights<16, 4, 3, 1>(cf3, sc3, wOut + 14848, tid, 256);
    prep_bias<4, 64>(sf0, bOut + 0,   tid, 256);
    prep_bias<64, 32>(sf1, bOut + 64,  tid, 256);
    prep_bias<32, 16>(sf2, bOut + 96,  tid, 256);
    prep_bias<16, 4>(sf3, bOut + 112, tid, 256);
}

// ---------------------------------------------------------------------------
// One KAN layer for a wave-private tile of 16 batch rows. hin holds xn.
// Bias is splatted into the C accumulator up front (lane owns one output
// column), so the WMMA accumulates it for free. Writeback applies the double
// tanh (outer tanh of this layer + inner tanh of the next kan_layer).
//   C layout: VGPR r -> row = (lane>>4)*8 + r, col = lane&15.
// ---------------------------------------------------------------------------
template <int DIN, int DOUT, int KSTEPS, int NTILES, int INSTRIDE, int OUTSTRIDE, bool TANH_OUT>
__device__ __forceinline__ void kan_stage(const float* hin, float* hout,
                                          const _Float16* w, const float* bias,
                                          int lane) {
    const int m    = lane & 15;
    const int half = lane >> 4;
    const int jcol = lane & 15;

    v8f acc[NTILES];
    static_for<NTILES>([&](auto nc) {
        constexpr int N = decltype(nc)::value;
        float bv;
        if constexpr (N * 16 + 15 < DOUT) bv = bias[N * 16 + jcol];
        else                              bv = (N * 16 + jcol < DOUT) ? bias[N * 16 + jcol] : 0.0f;
        static_for<8>([&](auto rc) { acc[N][decltype(rc)::value] = bv; });
    });

    static_for<KSTEPS>([&](auto kc) {
        constexpr int KS = decltype(kc)::value;
        v16h a = build_a<DIN, INSTRIDE, KS>(hin, m, half);
        static_for<NTILES>([&](auto nc) {
            constexpr int N = decltype(nc)::value;
            const v16h* bp = reinterpret_cast<const v16h*>(
                w + (size_t)((N * KSTEPS + KS) * 32 + lane) * 16);
            acc[N] = __builtin_amdgcn_wmma_f32_16x16x32_f16(
                false, a, false, *bp, (short)0, acc[N], false, false);
        });
    });

    static_for<NTILES>([&](auto nc) {
        constexpr int N = decltype(nc)::value;
        int j = N * 16 + jcol;
        static_for<8>([&](auto rc) {
            constexpr int R = decltype(rc)::value;
            int row = half * 8 + R;
            if (j < DOUT) {
                float z = acc[N][R];
                if (TANH_OUT) z = fast_tanh(fast_tanh(z));
                hout[row * OUTSTRIDE + j] = z;
            }
        });
    });
}

// ---------------------------------------------------------------------------
// Fused network: 4 KAN layers + Dirac gamma-sum epilogue.
// Block = 256 threads = 8 waves; each wave owns 16 batch rows -> 128 rows.
// Dynamic LDS: 32KB weights + 512B bias + 8 * 6144B activations = 82432B
// (well under CDNA5's 320KB/WGP).
// ---------------------------------------------------------------------------
__global__ void __launch_bounds__(256)
nkat_dirac_kan_kernel(const float* __restrict__ x,
                      const float* __restrict__ theta_log,
                      const float* __restrict__ nre,
                      const float* __restrict__ nim,
                      const uint4* __restrict__ wsrc,   // 2080 * 16B = weights+bias
                      float* __restrict__ out, int B) {
    extern __shared__ char smem[];
    _Float16* sW    = (_Float16*)smem;
    float*    sBias = (float*)(smem + 32768);

    // Bulk coalesced fill of weights+bias from the prepped ws image.
    {
        uint4* dst = (uint4*)smem;
        for (int i = threadIdx.x; i < 2080; i += 256) dst[i] = wsrc[i];
    }
    __syncthreads();

    const int tid     = threadIdx.x;
    const int w       = tid >> 5;
    const int lane    = tid & 31;
    const int rowBase = blockIdx.x * 128 + w * 16;

    float* hA = (float*)(smem + 33280 + w * 6144);          // stride 32
    float* hB = (float*)(smem + 33280 + w * 6144 + 2048);   // stride 64

    // Early prefetch of the noise stream consumed in the epilogue.
    __builtin_prefetch(&nre[(size_t)rowBase * 4], 0, 0);
    __builtin_prefetch(&nim[(size_t)rowBase * 4], 0, 0);

    // Stage in x with the layer-0 inner tanh applied (hbuf always holds xn).
    for (int idx = lane; idx < 64; idx += 32) {
        int row = idx >> 2, col = idx & 3;
        int gr  = rowBase + row;
        hA[row * 32 + col] = (gr < B) ? fast_tanh(x[(size_t)gr * 4 + col]) : 0.0f;
    }

    // Fragment offsets (f16 elements): L0 @0, L1 @2048, L2 @12288, L3 @14848
    kan_stage<4, 64, 1, 4, 32, 64, true >(hA, hB, sW + 0,     sBias + 0,   lane);
    __syncthreads();
    kan_stage<64, 32, 10, 2, 64, 32, true >(hB, hA, sW + 2048,  sBias + 64,  lane);
    __syncthreads();
    kan_stage<32, 16, 5, 1, 32, 64, true >(hA, hB, sW + 12288, sBias + 96,  lane);
    __syncthreads();
    kan_stage<16, 4, 3, 1, 64, 32, false>(hB, hA, sW + 14848, sBias + 112, lane);
    __syncthreads();

    // Dirac epilogue: GAMMA_SUM nonzero blocks:
    //   top-right  [[2, 1-i],[1+i, 0]]   (acts on k2,k3)
    //   bottom-left[[0,-1+i],[-1-i, 2]]  (acts on k0,k1)
    const float theta = __expf(theta_log[0]);
    const float ns    = theta * 1e-4f * 0.70710678118654752f;

    for (int idx = lane; idx < 64; idx += 32) {
        int row = idx >> 2, a = idx & 3;
        int gr  = rowBase + row;
        if (gr < B) {
            float k0 = hA[row * 32 + 0];
            float k1 = hA[row * 32 + 1];
            float k2 = hA[row * 32 + 2];
            float k3 = hA[row * 32 + 3];
            float re, im;
            if      (a == 0) { re = 2.0f * k2 + k3; im = -k3; }
            else if (a == 1) { re = k2;             im =  k2; }
            else if (a == 2) { re = -k1;            im =  k1; }
            else             { re = 2.0f * k1 - k0; im = -k0; }
            size_t e = (size_t)gr * 4 + a;
            out[e * 2 + 0] = re + ns * nre[e];
            out[e * 2 + 1] = im + ns * nim[e];
        }
    }

    if (blockIdx.x == 0 && tid == 0) out[(size_t)B * 8] = theta;
}

extern "C" void kernel_launch(void* const* d_in, const int* in_sizes, int n_in,
                              void* d_out, int out_size, void* d_ws, size_t ws_size,
                              hipStream_t stream) {
    // setup_inputs() dict order:
    // 0:x 1:theta_log 2:noise_re 3:noise_im 4:c0 5:s0 6:h0 7:c1 8:s1 9:h1
    // 10:c2 11:s2 12:h2 13:c3 14:s3 15:h3
    const float* x   = (const float*)d_in[0];
    const float* th  = (const float*)d_in[1];
    const float* nre = (const float*)d_in[2];
    const float* nim = (const float*)d_in[3];

    _Float16* wOut = (_Float16*)d_ws;
    float*    bOut = (float*)((char*)d_ws + 32768);

    nkat_prep_kernel<<<dim3(1), dim3(256), 0, stream>>>(
        (const float*)d_in[4],  (const float*)d_in[5],  (const float*)d_in[6],
        (const float*)d_in[7],  (const float*)d_in[8],  (const float*)d_in[9],
        (const float*)d_in[10], (const float*)d_in[11], (const float*)d_in[12],
        (const float*)d_in[13], (const float*)d_in[14], (const float*)d_in[15],
        wOut, bOut);

    const int B = in_sizes[0] / 4;
    dim3 grid((B + 127) / 128), block(256);
    const size_t smemBytes = 33280 + 8 * 6144;   // 82432
    nkat_dirac_kan_kernel<<<grid, block, smemBytes, stream>>>(
        x, th, nre, nim, (const uint4*)d_ws, (float*)d_out, B);
}
